// CFConv_6932077216272
// MI455X (gfx1250) — matrix operand order repaired and checked by reading, not compiled, not run on hardware
//
#include <hip/hip_runtime.h>

typedef __attribute__((ext_vector_type(16))) _Float16 v16h;
typedef __attribute__((ext_vector_type(8)))  _Float16 v8h;
typedef __attribute__((ext_vector_type(4)))  _Float16 v4h;
typedef __attribute__((ext_vector_type(2)))  _Float16 v2h;
typedef __attribute__((ext_vector_type(8)))  float    v8f;

#define B_   32
#define NA_  128
#define NN_  64
#define NF_  128   // n_in = n_filters = n_out
#define NG_  64    // n_gauss
#define CUTOFF_ 5.0f
#define LOG2F_  0.6931471805599453f
#define PI_     3.14159265358979323846f

// shifted softplus: log(1+exp(v)) - log(2); hardware v_exp_f32/v_log_f32,
// arg of log is in (1,2] so fast-log precision is fine.
__device__ __forceinline__ float ssp(float v) {
    float t = __expf(-fabsf(v));
    return fmaxf(v, 0.0f) + __logf(1.0f + t) - LOG2F_;
}

__device__ __forceinline__ v16h frag_from_rowmajor(const _Float16* __restrict__ row,
                                                   int k0, int h) {
    // 16-bit A-fragment: element i holds K = k0 + 16*h + i
    v8h lo = *(const v8h*)&row[k0 + 16*h];
    v8h hi = *(const v8h*)&row[k0 + 16*h + 8];
    return __builtin_shufflevector(lo, hi, 0,1,2,3,4,5,6,7,8,9,10,11,12,13,14,15);
}

// ---------- one-shot packing kernels ----------

// f32 -> f16 elementwise, 8 elements/thread (b128 in, b128 out)
__global__ __launch_bounds__(256)
void k_cvt_h8(const float* __restrict__ in, _Float16* __restrict__ out) {
    const size_t i = ((size_t)blockIdx.x * 256 + threadIdx.x) * 8;
    float4 a = *(const float4*)(in + i);
    float4 b = *(const float4*)(in + i + 4);
    v8h o = {(_Float16)a.x,(_Float16)a.y,(_Float16)a.z,(_Float16)a.w,
             (_Float16)b.x,(_Float16)b.y,(_Float16)b.z,(_Float16)b.w};
    *(v8h*)(out + i) = o;
}

// Pack W(K,N) f32 row-major into WMMA B-fragment order, f16:
// out[(((nt*nk + kc)*32 + lane)*16 + e]  with  k = kc*32 + 16*(lane>>4) + e,
// col = nt*16 + (lane&15)   (matches b[i] layout: K = k0 + 16*h + i)
__global__ __launch_bounds__(256)
void k_pack_b(const float* __restrict__ W, _Float16* __restrict__ out, int N, int nk) {
    const int i    = blockIdx.x * 256 + threadIdx.x;
    const int e    = i & 15;
    const int frag = i >> 4;
    const int lane = frag & 31;
    const int t    = frag >> 5;          // nt*nk + kc
    const int kc   = t % nk;
    const int nt   = t / nk;
    const int k    = kc * 32 + ((lane >> 4) << 4) + e;
    const int col  = (nt << 4) + (lane & 15);
    out[i] = (_Float16)W[(size_t)k * N + col];
}

// ---------- GEMM tile: f16 row-major A (global) x packed-f16 B ----------
__device__ __forceinline__ v8f wmma_tile_hp(const _Float16* __restrict__ A, int lda, int row0,
                                            const _Float16* __restrict__ Bp, int nk, int nt) {
    const int lane = threadIdx.x & 31;
    const int m = lane & 15, h = lane >> 4;
    v8f acc = {0.f,0.f,0.f,0.f,0.f,0.f,0.f,0.f};
#pragma unroll
    for (int kc = 0; kc < 4; ++kc) {     // K = 128 for both global GEMMs
        v16h a  = frag_from_rowmajor(&A[(size_t)(row0+m)*lda], kc*32, h);
        v16h bb = *(const v16h*)&Bp[((size_t)(nt*nk + kc) * 32 + lane) * 16];
        acc = __builtin_amdgcn_wmma_f32_16x16x32_f16(false, a, false, bb,
                                                     (short)0, acc, false, false);
    }
    return acc;
}

// Kernel 1: yh(4096,128) = xh @ W_in2f  (f16 out, no bias)
__global__ __launch_bounds__(256)
void k_in2f(const _Float16* __restrict__ xh, const _Float16* __restrict__ Wp,
            _Float16* __restrict__ yh) {
    const int row0 = blockIdx.x * 16;
    const int nt   = threadIdx.x >> 5;            // 8 waves cover 128 cols
    v8f acc = wmma_tile_hp(xh, NF_, row0, Wp, 4, nt);
    const int lane = threadIdx.x & 31;
    const int n = lane & 15, h = lane >> 4;
#pragma unroll
    for (int r = 0; r < 8; ++r)
        yh[(size_t)(row0 + r + 8*h)*NF_ + nt*16 + n] = (_Float16)acc[r];
}

// Kernel 3: out = ssp(acch @ W_out + b_out)  (f32 out)
__global__ __launch_bounds__(256)
void k_f2out(const _Float16* __restrict__ acch, const _Float16* __restrict__ Wp,
             const float* __restrict__ bout, float* __restrict__ out) {
    const int row0 = blockIdx.x * 16;
    const int nt   = threadIdx.x >> 5;
    v8f acc = wmma_tile_hp(acch, NF_, row0, Wp, 4, nt);
    const int lane = threadIdx.x & 31;
    const int n = lane & 15, h = lane >> 4;
#pragma unroll
    for (int r = 0; r < 8; ++r)
        out[(size_t)(row0 + r + 8*h)*NF_ + nt*16 + n] = ssp(acc[r] + bout[nt*16 + n]);
}

// Kernel 2: fused filter-net + cutoff + async neighbor gather + reduction.
__global__ __launch_bounds__(256)
void k_filter_gather(const float* __restrict__ f_ij, const float* __restrict__ r_ij,
                     const int*   __restrict__ nbidx, const float* __restrict__ mask,
                     const _Float16* __restrict__ Wf1p, const float* __restrict__ bf1,
                     const _Float16* __restrict__ Wf2p, const float* __restrict__ bf2,
                     const _Float16* __restrict__ yh,  _Float16* __restrict__ acch) {
    __shared__ __align__(16) _Float16 sF[NN_][NG_ + 8];    // 144B row stride
    __shared__ __align__(16) _Float16 sH[NN_][NF_ + 8];    // 272B row stride
    __shared__ __align__(16) _Float16 sW[NN_][NF_ + 4];    // final filter
    __shared__ __align__(16) _Float16 sY[NN_][NF_];        // async-gathered neighbors
    __shared__ float sC[NN_];
    __shared__ int   sNb[NN_];
    __shared__ float sRed[4][NF_/2][2];                    // [slice][fpair][j]

    const int ba  = blockIdx.x;               // flattened (b, a)
    const int b   = ba >> 7;
    const int tid = threadIdx.x;
    const int wave = tid >> 5, lane = tid & 31;
    const int m = lane & 15, h = lane >> 4;

    // ---- stage f_ij[b,a] (64x64 f32) into LDS as f16, vectorized ----
    {
        const float* fij = f_ij + (size_t)ba * NN_ * NG_;
        const float4* f4 = (const float4*)fij + (size_t)tid * 4;   // 16 floats/thread
        const int row = tid >> 2, cb = (tid & 3) * 16;
#pragma unroll
        for (int q = 0; q < 4; ++q) {
            float4 v = f4[q];
            v4h o = {(_Float16)v.x,(_Float16)v.y,(_Float16)v.z,(_Float16)v.w};
            *(v4h*)&sF[row][cb + q*4] = o;
        }
    }

    if (tid < NN_) {
        float r = r_ij[(size_t)ba * NN_ + tid];
        float c = 0.5f * (__cosf(r * (PI_ / CUTOFF_)) + 1.0f);
        c = (r < CUTOFF_) ? c : 0.0f;
        sC[tid]  = c * mask[(size_t)ba * NN_ + tid];
        sNb[tid] = nbidx[(size_t)ba * NN_ + tid];
    }
    __syncthreads();

    // ---- async prefetch of gathered neighbor rows into LDS (overlaps GEMMs) ----
    // 64 rows x 256B = 1024 x b128 transfers; 4 per thread.
#pragma unroll
    for (int k = 0; k < 4; ++k) {
        const int j   = tid + 256 * k;        // transfer index
        const int n   = j >> 4;               // neighbor row
        const int seg = j & 15;               // 16B segment within row
        const int idx = sNb[n];
        const _Float16* gp = yh + ((size_t)(b * NA_) + idx) * NF_ + seg * 8;
        unsigned long long ga = (unsigned long long)(uintptr_t)gp;
        unsigned int lo = (unsigned int)(uintptr_t)&sY[n][seg * 8];
        asm volatile("global_load_async_to_lds_b128 %0, %1, off"
                     :: "v"(lo), "v"(ga) : "memory");
    }

    // ---- GEMM1: H = ssp(f_ij @ Wf1 + bf1), 64x128, K=64 ----
    {
        const int row0 = (wave & 3) * 16;
#pragma unroll
        for (int j = 0; j < 4; ++j) {
            const int nt = ((wave >> 2) << 2) + j;
            v8f acc = {0.f,0.f,0.f,0.f,0.f,0.f,0.f,0.f};
#pragma unroll
            for (int kc = 0; kc < 2; ++kc) {
                v16h a  = frag_from_rowmajor(&sF[row0 + m][0], kc*32, h);
                v16h bb = *(const v16h*)&Wf1p[((size_t)(nt*2 + kc)*32 + lane)*16];
                acc = __builtin_amdgcn_wmma_f32_16x16x32_f16(false, a, false, bb,
                                                             (short)0, acc, false, false);
            }
            const float bias = bf1[nt*16 + m];
#pragma unroll
            for (int r = 0; r < 8; ++r)
                sH[row0 + r + 8*h][nt*16 + m] = (_Float16)ssp(acc[r] + bias);
        }
    }
    __syncthreads();

    // ---- GEMM2: Wfin = (H @ Wf2 + bf2) * C[row], 64x128, K=128 ----
    {
        const int row0 = (wave & 3) * 16;
#pragma unroll
        for (int j = 0; j < 4; ++j) {
            const int nt = ((wave >> 2) << 2) + j;
            v8f acc = {0.f,0.f,0.f,0.f,0.f,0.f,0.f,0.f};
#pragma unroll
            for (int kc = 0; kc < 4; ++kc) {
                v16h a  = frag_from_rowmajor(&sH[row0 + m][0], kc*32, h);
                v16h bb = *(const v16h*)&Wf2p[((size_t)(nt*4 + kc)*32 + lane)*16];
                acc = __builtin_amdgcn_wmma_f32_16x16x32_f16(false, a, false, bb,
                                                             (short)0, acc, false, false);
            }
            const float bias = bf2[nt*16 + m];
#pragma unroll
            for (int r = 0; r < 8; ++r) {
                const int row = row0 + r + 8*h;
                sW[row][nt*16 + m] = (_Float16)((acc[r] + bias) * sC[row]);
            }
        }
    }

    // ---- wait for async gather, then reduce over neighbors ----
    asm volatile("s_wait_asynccnt 0" ::: "memory");
    __syncthreads();
    {
        // thread owns feature pair f0=2*p over a 16-neighbor slice; dword LDS loads
        const int p     = tid & 63;           // feature pair index
        const int slice = tid >> 6;           // 4 slices of 16 neighbors
        const int f0    = p * 2;
        float p0 = 0.f, p1 = 0.f;
#pragma unroll
        for (int t = 0; t < 16; ++t) {
            const int n = slice * 16 + t;
            v2h w  = *(const v2h*)&sW[n][f0];
            v2h yv = *(const v2h*)&sY[n][f0];
            p0 += (float)w[0] * (float)yv[0];   // v_fma_mix_f32
            p1 += (float)w[1] * (float)yv[1];
        }
        sRed[slice][p][0] = p0;
        sRed[slice][p][1] = p1;
    }
    __syncthreads();
    if (tid < NF_) {
        const int p = tid >> 1, j = tid & 1;
        float s = sRed[0][p][j] + sRed[1][p][j] + sRed[2][p][j] + sRed[3][p][j];
        acch[(size_t)ba * NF_ + tid] = (_Float16)s;
    }
}

extern "C" void kernel_launch(void* const* d_in, const int* in_sizes, int n_in,
                              void* d_out, int out_size, void* d_ws, size_t ws_size,
                              hipStream_t stream) {
    const float* x      = (const float*)d_in[0];
    const float* r_ij   = (const float*)d_in[1];
    const int*   nbr    = (const int*)  d_in[2];
    const float* mask   = (const float*)d_in[3];
    const float* f_ij   = (const float*)d_in[4];
    const float* W_in2f = (const float*)d_in[5];
    const float* Wf1    = (const float*)d_in[6];
    const float* bf1    = (const float*)d_in[7];
    const float* Wf2    = (const float*)d_in[8];
    const float* bf2    = (const float*)d_in[9];
    const float* W_out  = (const float*)d_in[10];
    const float* b_out  = (const float*)d_in[11];
    float* out = (float*)d_out;

    // workspace layout (f16 unless noted)
    _Float16* yh   = (_Float16*)d_ws;                        // 4096*128   = 1MB
    _Float16* acch = yh   + (size_t)B_*NA_*NF_;              // 1MB
    _Float16* xh   = acch + (size_t)B_*NA_*NF_;              // 1MB
    _Float16* Wip  = xh   + (size_t)B_*NA_*NF_;              // 128*128 packed
    _Float16* W1p  = Wip  + (size_t)NF_*NF_;                 // 64*128 packed
    _Float16* W2p  = W1p  + (size_t)NG_*NF_;                 // 128*128 packed
    _Float16* Wop  = W2p  + (size_t)NF_*NF_;                 // 128*128 packed

    k_cvt_h8<<<(B_*NA_*NF_)/(256*8), 256, 0, stream>>>(x, xh);
    k_pack_b<<<(NF_*NF_)/256,        256, 0, stream>>>(W_in2f, Wip, NF_, 4);
    k_pack_b<<<(NG_*NF_)/256,        256, 0, stream>>>(Wf1,    W1p, NF_, 2);
    k_pack_b<<<(NF_*NF_)/256,        256, 0, stream>>>(Wf2,    W2p, NF_, 4);
    k_pack_b<<<(NF_*NF_)/256,        256, 0, stream>>>(W_out,  Wop, NF_, 4);

    k_in2f         <<<(B_*NA_)/16, 256, 0, stream>>>(xh, Wip, yh);
    k_filter_gather<<<B_*NA_,      256, 0, stream>>>(f_ij, r_ij, nbr, mask,
                                                     W1p, bf1, W2p, bf2, yh, acch);
    k_f2out        <<<(B_*NA_)/16, 256, 0, stream>>>(acch, Wop, b_out, out);
}
